// Layer_6493990551861
// MI455X (gfx1250) — matrix operand order repaired and checked by reading, not compiled, not run on hardware
//
#include <hip/hip_runtime.h>
#include <math.h>

#define DEV __device__ __forceinline__

typedef __attribute__((ext_vector_type(16))) __bf16 v16bf;
typedef __attribute__((ext_vector_type(8)))  __bf16 v8bf;
typedef __attribute__((ext_vector_type(8)))  float  v8f;

static constexpr int kB  = 2;
static constexpr int kS  = 2048;
static constexpr int kH  = 1024;
static constexpr int kNH = 16;
static constexpr int kHD = 64;
static constexpr int kFF = 4096;
static constexpr int kM  = kB * kS;   // 4096 rows total

// ---------- fragment helpers (CDNA5 wave32 WMMA layouts, CDNA5 ISA) ----------

DEV v16bf make16(v8bf lo, v8bf hi) {
    v16bf r;
#pragma unroll
    for (int i = 0; i < 8; ++i) { r[i] = lo[i]; r[i + 8] = hi[i]; }
    return r;
}

// A matrix 16x32 bf16: lane m = l%16; khalf = (l<16)?0:8; elements 0..7 -> K=khalf..+7,
// elements 8..15 -> K=16+khalf..+7.  Two contiguous 16B loads per lane.
DEV v16bf load_a_frag(const __bf16* base, int lda) {
    const int lane = threadIdx.x & 31;
    const __bf16* p = base + (size_t)(lane & 15) * lda + ((lane & 16) ? 8 : 0);
    return make16(*(const v8bf*)p, *(const v8bf*)(p + 16));
}

// B matrix 32x16 bf16 (source given transposed [N][K]): lane n = l%16;
// khalf = (l<16)?0:8; elements 0..7 -> K=khalf..+7, elements 8..15 -> K=16+khalf..+7.
DEV v16bf load_b_frag(const __bf16* base, int ldb) {
    const int lane = threadIdx.x & 31;
    const __bf16* p = base + (size_t)(lane & 15) * ldb + ((lane & 16) ? 8 : 0);
    return make16(*(const v8bf*)p, *(const v8bf*)(p + 16));
}

DEV v8f wmma_bf16(v16bf a, v16bf b, v8f c) {
    return __builtin_amdgcn_wmma_f32_16x16x32_bf16(false, a, false, b, (short)0, c,
                                                   false, false);
}

DEV v8f vzero() {
    v8f z;
#pragma unroll
    for (int i = 0; i < 8; ++i) z[i] = 0.0f;
    return z;
}

// ---------- conversion kernels ----------

__global__ __launch_bounds__(256) void cvt_bf16_k(const float* __restrict__ in,
                                                  __bf16* __restrict__ out, int n) {
    int i = blockIdx.x * 256 + threadIdx.x;
    if (i < n) out[i] = (__bf16)in[i];
}

// W[K][N] f32  ->  Wt[N][K] bf16  (tiled via LDS)
__global__ __launch_bounds__(256) void transpose_bf16_k(const float* __restrict__ W,
                                                        __bf16* __restrict__ Wt,
                                                        int K, int N) {
    __shared__ float tile[32][33];
    const int n0 = blockIdx.x * 32;
    const int k0 = blockIdx.y * 32;
    for (int i = threadIdx.y; i < 32; i += 8)
        tile[i][threadIdx.x] = W[(size_t)(k0 + i) * N + n0 + threadIdx.x];
    __syncthreads();
    for (int i = threadIdx.y; i < 32; i += 8)
        Wt[(size_t)(n0 + i) * K + k0 + threadIdx.x] = (__bf16)tile[threadIdx.x][i];
}

// ---------- GEMM: C[M,N] = A[M,K] x B[K,N] + bias, B given transposed [N][K] ----------
// One wave (block of 32) computes a 32x64 tile: 2 A fragments x 4 B fragments ->
// 8 WMMAs per 32-deep k-step, so each B fragment feeds two WMMAs.
// EPI: 0=f32, 1=bf16, 2=bf16+gelu, 3=bf16 stored V-transposed as [b][head][d][s].

template <int EPI>
__global__ __launch_bounds__(32) void gemm_bf16_k(const __bf16* __restrict__ A,
                                                  const __bf16* __restrict__ Bt,
                                                  const float* __restrict__ bias,
                                                  void* __restrict__ Cout,
                                                  int Nd, int Kd) {
    const int m0 = blockIdx.x * 32;
    const int n0 = blockIdx.y * 64;
    v8f acc[2][4];
#pragma unroll
    for (int r = 0; r < 2; ++r)
#pragma unroll
        for (int t = 0; t < 4; ++t) acc[r][t] = vzero();

    for (int k0 = 0; k0 < Kd; k0 += 32) {
        v16bf a0 = load_a_frag(A + (size_t)m0 * Kd + k0, Kd);
        v16bf a1 = load_a_frag(A + (size_t)(m0 + 16) * Kd + k0, Kd);
        if (k0 + 32 < Kd) {  // hint the next A k-block into cache (global_prefetch_b8)
            __builtin_prefetch(A + (size_t)m0 * Kd + k0 + 32, 0, 1);
            __builtin_prefetch(A + (size_t)(m0 + 16) * Kd + k0 + 32, 0, 1);
        }
#pragma unroll
        for (int t = 0; t < 4; ++t) {
            v16bf b = load_b_frag(Bt + (size_t)(n0 + t * 16) * Kd + k0, Kd);
            acc[0][t] = wmma_bf16(a0, b, acc[0][t]);
            acc[1][t] = wmma_bf16(a1, b, acc[1][t]);
        }
    }

    const int lane = threadIdx.x & 31;
    const int n    = lane & 15;
    const int mh   = (lane & 16) ? 8 : 0;   // C layout: lanes 16-31 hold rows v+8
#pragma unroll
    for (int r = 0; r < 2; ++r) {
#pragma unroll
        for (int t = 0; t < 4; ++t) {
            const int col = n0 + t * 16 + n;
            const float bvv = bias[col];
#pragma unroll
            for (int v = 0; v < 8; ++v) {
                const int m = m0 + r * 16 + v + mh;
                float x = acc[r][t][v] + bvv;
                if (EPI == 2) x = 0.5f * x * (1.0f + erff(x * 0.70710678118654752f));
                if (EPI == 0) {
                    ((float*)Cout)[(size_t)m * Nd + col] = x;
                } else if (EPI == 3) {
                    const int bb = m / kS, srow = m % kS;
                    const int head = col / kHD, d = col % kHD;
                    ((__bf16*)Cout)[(((size_t)bb * kNH + head) * kHD + d) * kS + srow] =
                        (__bf16)x;
                } else {
                    ((__bf16*)Cout)[(size_t)m * Nd + col] = (__bf16)x;
                }
            }
        }
    }
}

// ---------- fused attention: one wave per (b*NH + h, 16-query tile) ----------
// qb/kb: [M][H] bf16 row-major; vtb: [b][h][d][s] bf16; ab out: [M][H] bf16.

__global__ __launch_bounds__(32) void attention_k(const __bf16* __restrict__ qb,
                                                  const __bf16* __restrict__ kb,
                                                  const __bf16* __restrict__ vtb,
                                                  const int* __restrict__ am,
                                                  __bf16* __restrict__ ab) {
    __shared__ __bf16 Pb[16 * 32];
    const int qt = blockIdx.x;
    const int bh = blockIdx.y;
    const int b  = bh / kNH;
    const int h  = bh % kNH;
    const int lane = threadIdx.x & 31;
    const int n    = lane & 15;
    const int mh   = (lane & 16) ? 8 : 0;

    const __bf16* Qbase = qb + (size_t)(b * kS + qt * 16) * kH + h * kHD;
    const __bf16* Kbase = kb + (size_t)(b * kS) * kH + h * kHD;
    const __bf16* Vbase = vtb + (size_t)bh * kHD * kS;
    const int* mrow = am + b * kS;

    // Q tile 16x64 as two A fragments (d chunks 0..31, 32..63), kept in VGPRs
    v16bf Aq0 = load_a_frag(Qbase, kH);
    v16bf Aq1 = load_a_frag(Qbase + 32, kH);

    float mrun[8], lrun[8];
    v8f o[4];
#pragma unroll
    for (int v = 0; v < 8; ++v) { mrun[v] = -__builtin_inff(); lrun[v] = 0.0f; }
#pragma unroll
    for (int t = 0; t < 4; ++t) o[t] = vzero();

    for (int k0 = 0; k0 < kS; k0 += 32) {
        // scores S(16x32) = Q(16x64) . K^T(64x32), scale 1/sqrt(64)
        v8f s0 = vzero(), s1 = vzero();
        s0 = wmma_bf16(Aq0, load_b_frag(Kbase + (size_t)k0 * kH, kH), s0);
        s0 = wmma_bf16(Aq1, load_b_frag(Kbase + (size_t)k0 * kH + 32, kH), s0);
        s1 = wmma_bf16(Aq0, load_b_frag(Kbase + (size_t)(k0 + 16) * kH, kH), s1);
        s1 = wmma_bf16(Aq1, load_b_frag(Kbase + (size_t)(k0 + 16) * kH + 32, kH), s1);

        const bool ok0 = (mrow[k0 + n] != 0);
        const bool ok1 = (mrow[k0 + 16 + n] != 0);

        float mnew[8];
#pragma unroll
        for (int v = 0; v < 8; ++v) {
            float a0 = ok0 ? s0[v] * 0.125f : -__builtin_inff();
            float a1 = ok1 ? s1[v] * 0.125f : -__builtin_inff();
            s0[v] = a0; s1[v] = a1;
            // row max across 16 lanes of this half-wave (row m lives in one half)
            float mx = fmaxf(a0, a1);
            mx = fmaxf(mx, __shfl_xor(mx, 1, 16));
            mx = fmaxf(mx, __shfl_xor(mx, 2, 16));
            mx = fmaxf(mx, __shfl_xor(mx, 4, 16));
            mx = fmaxf(mx, __shfl_xor(mx, 8, 16));
            mnew[v] = fmaxf(mrun[v], mx);
        }
#pragma unroll
        for (int v = 0; v < 8; ++v) {
            const float alpha = __expf(mrun[v] - mnew[v]);
            const float p0 = __expf(s0[v] - mnew[v]);
            const float p1 = __expf(s1[v] - mnew[v]);
            s0[v] = p0; s1[v] = p1;
            float rs = p0 + p1;
            rs += __shfl_xor(rs, 1, 16);
            rs += __shfl_xor(rs, 2, 16);
            rs += __shfl_xor(rs, 4, 16);
            rs += __shfl_xor(rs, 8, 16);
            lrun[v] = lrun[v] * alpha + rs;
            mrun[v] = mnew[v];
            o[0][v] *= alpha; o[1][v] *= alpha; o[2][v] *= alpha; o[3][v] *= alpha;
        }
        // reshape P from C-layout (f32) to A-layout (bf16) through LDS
        __syncthreads();
#pragma unroll
        for (int v = 0; v < 8; ++v) {
            Pb[(v + mh) * 32 + n]      = (__bf16)s0[v];
            Pb[(v + mh) * 32 + 16 + n] = (__bf16)s1[v];
        }
        __syncthreads();
        v16bf Ap = load_a_frag(&Pb[0], 32);
        // O(16x64) += P(16x32) . V(32x64); V^T layout makes B-frags contiguous
#pragma unroll
        for (int dt = 0; dt < 4; ++dt) {
            v16bf bv = load_b_frag(Vbase + (size_t)(dt * 16) * kS + k0, kS);
            o[dt] = wmma_bf16(Ap, bv, o[dt]);
        }
    }

    __bf16* Obase = ab + (size_t)(b * kS + qt * 16) * kH + h * kHD;
#pragma unroll
    for (int dt = 0; dt < 4; ++dt)
#pragma unroll
        for (int v = 0; v < 8; ++v)
            Obase[(size_t)(v + mh) * kH + dt * 16 + n] = (__bf16)(o[dt][v] / lrun[v]);
}

// ---------- layernorm over H=1024 per token ----------

template <typename OutT>
__global__ __launch_bounds__(256) void layernorm_k(const float* __restrict__ in,
                                                   const float* __restrict__ g,
                                                   const float* __restrict__ bta,
                                                   OutT* __restrict__ out) {
    __shared__ float red[256];
    const int row = blockIdx.x;
    const float* x = in + (size_t)row * kH;
    const int tid = threadIdx.x;

    float s = 0.0f;
    for (int i = tid; i < kH; i += 256) s += x[i];
    red[tid] = s; __syncthreads();
    for (int st = 128; st > 0; st >>= 1) {
        if (tid < st) red[tid] += red[tid + st];
        __syncthreads();
    }
    const float mean = red[0] * (1.0f / kH);
    __syncthreads();

    float vs = 0.0f;
    for (int i = tid; i < kH; i += 256) { float d = x[i] - mean; vs += d * d; }
    red[tid] = vs; __syncthreads();
    for (int st = 128; st > 0; st >>= 1) {
        if (tid < st) red[tid] += red[tid + st];
        __syncthreads();
    }
    const float inv = rsqrtf(red[0] * (1.0f / kH) + 1e-12f);

    for (int i = tid; i < kH; i += 256)
        out[(size_t)row * kH + i] = (OutT)(g[i] * (x[i] - mean) * inv + bta[i]);
}

// ---------- driver ----------

extern "C" void kernel_launch(void* const* d_in, const int* in_sizes, int n_in,
                              void* d_out, int out_size, void* d_ws, size_t ws_size,
                              hipStream_t stream) {
    (void)in_sizes; (void)n_in; (void)out_size; (void)ws_size;

    const float* x    = (const float*)d_in[0];
    const int*   amsk = (const int*)d_in[1];
    const float* Wq   = (const float*)d_in[2];
    const float* bq   = (const float*)d_in[3];
    const float* Wk   = (const float*)d_in[4];
    const float* bk   = (const float*)d_in[5];
    const float* Wv   = (const float*)d_in[6];
    const float* bv   = (const float*)d_in[7];
    const float* Wo   = (const float*)d_in[8];
    const float* bo   = (const float*)d_in[9];
    const float* ln1g = (const float*)d_in[10];
    const float* ln1b = (const float*)d_in[11];
    const float* W1   = (const float*)d_in[12];
    const float* b1   = (const float*)d_in[13];
    const float* W2   = (const float*)d_in[14];
    const float* b2   = (const float*)d_in[15];
    const float* ln2g = (const float*)d_in[16];
    const float* ln2b = (const float*)d_in[17];

    char* ws = (char*)d_ws;
    const size_t MB = 1024ull * 1024ull;
    __bf16* xb   = (__bf16*)(ws + 0 * MB);    // 8 MB
    __bf16* wqT  = (__bf16*)(ws + 8 * MB);    // 2 MB
    __bf16* wkT  = (__bf16*)(ws + 10 * MB);   // 2 MB
    __bf16* wvT  = (__bf16*)(ws + 12 * MB);   // 2 MB
    __bf16* woT  = (__bf16*)(ws + 14 * MB);   // 2 MB
    __bf16* w1T  = (__bf16*)(ws + 16 * MB);   // 8 MB
    __bf16* w2T  = (__bf16*)(ws + 24 * MB);   // 8 MB
    __bf16* qbb  = (__bf16*)(ws + 32 * MB);   // 8 MB
    __bf16* kbb  = (__bf16*)(ws + 40 * MB);   // 8 MB
    __bf16* vtb  = (__bf16*)(ws + 48 * MB);   // 8 MB  V^T per head [b][h][d][s]
    __bf16* abf  = (__bf16*)(ws + 56 * MB);   // 8 MB  attention output
    __bf16* l1b  = (__bf16*)(ws + 64 * MB);   // 8 MB  LN1 out (bf16)
    __bf16* h1b  = (__bf16*)(ws + 72 * MB);   // 32 MB FFN hidden (bf16)
    float*  tmpf = (float*)(ws + 104 * MB);   // 16 MB f32 staging

    const dim3 tb(32, 8);

    cvt_bf16_k<<<dim3((kM * kH) / 256), 256, 0, stream>>>(x, xb, kM * kH);
    transpose_bf16_k<<<dim3(kH / 32, kH / 32), tb, 0, stream>>>(Wq, wqT, kH, kH);
    transpose_bf16_k<<<dim3(kH / 32, kH / 32), tb, 0, stream>>>(Wk, wkT, kH, kH);
    transpose_bf16_k<<<dim3(kH / 32, kH / 32), tb, 0, stream>>>(Wv, wvT, kH, kH);
    transpose_bf16_k<<<dim3(kH / 32, kH / 32), tb, 0, stream>>>(Wo, woT, kH, kH);
    transpose_bf16_k<<<dim3(kFF / 32, kH / 32), tb, 0, stream>>>(W1, w1T, kH, kFF);
    transpose_bf16_k<<<dim3(kH / 32, kFF / 32), tb, 0, stream>>>(W2, w2T, kFF, kH);

    // QKV projections (32x64 tile per wave)
    gemm_bf16_k<1><<<dim3(kM / 32, kH / 64), 32, 0, stream>>>(xb, wqT, bq, qbb, kH, kH);
    gemm_bf16_k<1><<<dim3(kM / 32, kH / 64), 32, 0, stream>>>(xb, wkT, bk, kbb, kH, kH);
    gemm_bf16_k<3><<<dim3(kM / 32, kH / 64), 32, 0, stream>>>(xb, wvT, bv, vtb, kH, kH);

    // fused softmax attention
    attention_k<<<dim3(kS / 16, kB * kNH), 32, 0, stream>>>(qbb, kbb, vtb, amsk, abf);

    // output projection + LN1
    gemm_bf16_k<0><<<dim3(kM / 32, kH / 64), 32, 0, stream>>>(abf, woT, bo, tmpf, kH, kH);
    layernorm_k<__bf16><<<kM, 256, 0, stream>>>(tmpf, ln1g, ln1b, l1b);

    // FFN with GELU(erf) + LN2 -> output (f32)
    gemm_bf16_k<2><<<dim3(kM / 32, kFF / 64), 32, 0, stream>>>(l1b, w1T, b1, h1b, kFF, kH);
    gemm_bf16_k<0><<<dim3(kM / 32, kH / 64), 32, 0, stream>>>(h1b, w2T, b2, tmpf, kH, kFF);
    layernorm_k<float><<<kM, 256, 0, stream>>>(tmpf, ln2g, ln2b, (float*)d_out);
}
